// SNNTokenMixingBlock_68633577390518
// MI455X (gfx1250) — matrix operand order repaired and checked by reading, not compiled, not run on hardware
//
#include <hip/hip_runtime.h>
#include <hip/hip_bf16.h>

#define T_ 4
#define B_ 16
#define C_ 256
#define H_ 32
#define W_ 32
#define HW_ 1024

typedef __attribute__((ext_vector_type(16))) __bf16 v16bf;
typedef __attribute__((ext_vector_type(8)))  float  v8f;
typedef __attribute__((ext_vector_type(8)))  unsigned short us8;
typedef __attribute__((ext_vector_type(4)))  float  f4;

__device__ __forceinline__ unsigned short f2bf(float f) {
  unsigned int u = __float_as_uint(f);
  u = u + 0x7FFFu + ((u >> 16) & 1u);      // round-to-nearest-even
  return (unsigned short)(u >> 16);
}

// --- WMMA fragment swizzle: LDS holds operands in per-lane register layout --
// A (16x32 bf16): lane = (k>>4)*16 + (m&15), elem i = k&15
// 16-row tiles stacked every 512 ushorts.
__device__ __forceinline__ int swzA(int m, int k) {
  return (m >> 4) * 512 + (((k >> 4) * 16 + (m & 15)) * 16)
         + (k & 15);
}
// B (32x16 bf16): lane = (k>>4)*16 + (n&15), elem i = k&15
// 16-col tiles stacked every 512 ushorts.
__device__ __forceinline__ int swzB(int k, int n) {
  return (n >> 4) * 512 + (((k >> 4) * 16 + (n & 15)) * 16) + (k & 15);
}

// ---------------------------------------------------------------------------
// Kernel A: pre_bn -> depthwise 3x3 conv -> dw_bn + residual -> LIF over T.
// One block per (b,c) plane; LIF membrane state in registers across T.
// Emits binary spikes as bf16 bits (0x0000 / 0x3F80).
// ---------------------------------------------------------------------------
__global__ __launch_bounds__(256) void lif_front_kernel(
    const float* __restrict__ fm,
    const float* __restrict__ pre_g, const float* __restrict__ pre_b,
    const float* __restrict__ pre_m, const float* __restrict__ pre_v,
    const float* __restrict__ dw_w,  const float* __restrict__ dw_b,
    const float* __restrict__ dbn_g, const float* __restrict__ dbn_b,
    const float* __restrict__ dbn_m, const float* __restrict__ dbn_v,
    unsigned short* __restrict__ xspk)
{
  __shared__ float lds[34 * 34];           // zero-padded plane
  const int bc  = blockIdx.x;
  const int b   = bc / C_, c = bc % C_;
  const int tid = threadIdx.x;

  const float sp = pre_g[c] * rsqrtf(pre_v[c] + 1e-5f);
  const float tp = pre_b[c] - pre_m[c] * sp;
  const float sd = dbn_g[c] * rsqrtf(dbn_v[c] + 1e-5f);
  const float td = dbn_b[c] - dbn_m[c] * sd;
  float wt[9];
#pragma unroll
  for (int k = 0; k < 9; ++k) wt[k] = dw_w[c * 9 + k];
  const float bias = dw_b[c];

  float vst[4] = {0.f, 0.f, 0.f, 0.f};     // LIF membrane, 4 pixels/thread

  for (int t = 0; t < T_; ++t) {
    const float* src = fm + (((size_t)t * B_ + b) * C_ + c) * HW_;
    for (int i = tid; i < 34 * 34; i += 256) lds[i] = 0.0f;
    __syncthreads();
#pragma unroll
    for (int i = 0; i < 4; ++i) {
      int p = tid + i * 256;
      int h = p >> 5, w = p & 31;
      lds[(h + 1) * 34 + (w + 1)] = src[p] * sp + tp;
    }
    __syncthreads();
    unsigned short* dst = xspk + (((size_t)t * B_ + b) * C_ + c) * HW_;
#pragma unroll
    for (int i = 0; i < 4; ++i) {
      int p = tid + i * 256;
      int h = p >> 5, w = p & 31;
      float acc = 0.f;
#pragma unroll
      for (int dy = 0; dy < 3; ++dy)
#pragma unroll
        for (int dx = 0; dx < 3; ++dx)
          acc = fmaf(lds[(h + dy) * 34 + (w + dx)], wt[dy * 3 + dx], acc);
      float y = (acc + bias) * sd + td + lds[(h + 1) * 34 + (w + 1)];
      float v = 0.5f * (vst[i] + y);       // v += (y - v)/tau, tau=2
      bool  s = (v >= 1.0f);
      vst[i]  = s ? 0.0f : v;              // hard reset
      dst[p]  = s ? (unsigned short)0x3F80 : (unsigned short)0;
    }
    __syncthreads();
  }
}

// ---------------------------------------------------------------------------
// Wave-level 32x32x32 bf16 GEMM on swizzled LDS operands: C = A @ B.
// Each fragment load is one contiguous 32B vector per lane (-> 2x ds_load_b128).
// ---------------------------------------------------------------------------
__device__ __forceinline__ void wave_gemm32s(const unsigned short* __restrict__ A,
                                             const unsigned short* __restrict__ Bm,
                                             float* __restrict__ Cout, int lane)
{
  const int half = lane >> 4, mn = lane & 15;
#pragma unroll
  for (int ti = 0; ti < 2; ++ti) {
    v16bf a = *(const v16bf*)&A[ti * 512 + lane * 16];
#pragma unroll
    for (int tj = 0; tj < 2; ++tj) {
      v16bf bb = *(const v16bf*)&Bm[tj * 512 + lane * 16];
      v8f cz = {0.f, 0.f, 0.f, 0.f, 0.f, 0.f, 0.f, 0.f};
      cz = __builtin_amdgcn_wmma_f32_16x16x32_bf16(false, a, false, bb,
                                                   (short)0, cz, false, false);
#pragma unroll
      for (int r = 0; r < 8; ++r)
        Cout[(ti * 16 + half * 8 + r) * 32 + tj * 16 + mn] = cz[r];
    }
  }
}

// ---------------------------------------------------------------------------
// Kernel B: SNNMlpBlock. One wave per (b,c). MODE is a template parameter so
// each instantiation is fully specialized (no runtime-branch register merges).
// MODE 0: Y = S @ W^T (mlp_w);  MODE 1: Y = W @ S (mlp_h).
// LIF states (2x 32x32) live in LDS across the T loop.
// ---------------------------------------------------------------------------
template <int MODE>
__global__ __launch_bounds__(32) void mlp_kernel(
    const unsigned short* __restrict__ xspk,
    const float* __restrict__ w1, const float* __restrict__ b1,
    const float* __restrict__ n1g, const float* __restrict__ n1b,
    const float* __restrict__ n1m, const float* __restrict__ n1v,
    const float* __restrict__ w2, const float* __restrict__ b2,
    const float* __restrict__ n2g, const float* __restrict__ n2b,
    const float* __restrict__ n2m, const float* __restrict__ n2v,
    unsigned short* __restrict__ out)
{
  __shared__ unsigned short ldsS[1024];    // spikes (A slot MODE0, B slot MODE1)
  __shared__ unsigned short ldsW1[1024];   // weights, swizzled for their slot
  __shared__ unsigned short ldsW2[1024];
  __shared__ float ldsY[1024];
  __shared__ float ldsH[1024];
  __shared__ float v1s[1024];
  __shared__ float v2s[1024];

  const int bc = blockIdx.x;
  const int b = bc / C_, c = bc % C_;
  const int lane = threadIdx.x;

  const float s1 = n1g[c] * rsqrtf(n1v[c] + 1e-5f);
  const float t1 = n1b[c] - n1m[c] * s1;
  const float s2 = n2g[c] * rsqrtf(n2v[c] + 1e-5f);
  const float t2 = n2b[c] - n2m[c] * s2;

  for (int j = 0; j < 32; ++j) {
    int idx = lane + j * 32;
    int r = idx >> 5, cc = idx & 31;
    if (MODE == 0) {  // weights in B slot: B[k=cc][n=r] = w[r][cc]
      ldsW1[swzB(cc, r)] = f2bf(w1[r * 32 + cc]);
      ldsW2[swzB(cc, r)] = f2bf(w2[r * 32 + cc]);
    } else {          // weights in A slot: A[m=r][k=cc] = w[r][cc]
      ldsW1[swzA(r, cc)] = f2bf(w1[r * 32 + cc]);
      ldsW2[swzA(r, cc)] = f2bf(w2[r * 32 + cc]);
    }
    v1s[idx] = 0.0f;
    v2s[idx] = 0.0f;
  }
  __syncthreads();

  for (int t = 0; t < T_; ++t) {
    const unsigned short* src = xspk + (((size_t)t * B_ + b) * C_ + c) * HW_;
#pragma unroll
    for (int j = 0; j < 4; ++j) {          // 4x ushort8 per lane, b128 loads
      int g = lane + j * 32;
      int h = g >> 2, w8 = (g & 3) * 8;
      us8 v = *(const us8*)&src[h * 32 + w8];
#pragma unroll
      for (int e = 0; e < 8; ++e) {
        int w = w8 + e;
        ldsS[(MODE == 0) ? swzA(h, w) : swzB(h, w)] = v[e];
      }
    }
    __syncthreads();

    if (MODE == 0) wave_gemm32s(ldsS, ldsW1, ldsY, lane);
    else           wave_gemm32s(ldsW1, ldsS, ldsY, lane);
    __syncthreads();

    for (int j = 0; j < 32; ++j) {         // bias + bn1 + LIF1
      int idx = lane + j * 32;
      int m = idx >> 5, n = idx & 31;
      float bias = (MODE == 0) ? b1[n] : b1[m];
      float y = (ldsY[idx] + bias) * s1 + t1;
      ldsH[idx] = y;                       // residual h
      float v = 0.5f * (v1s[idx] + y);
      bool  s = (v >= 1.0f);
      v1s[idx] = s ? 0.0f : v;
      unsigned short sv = s ? (unsigned short)0x3F80 : (unsigned short)0;
      ldsS[(MODE == 0) ? swzA(m, n) : swzB(m, n)] = sv;
    }
    __syncthreads();

    if (MODE == 0) wave_gemm32s(ldsS, ldsW2, ldsY, lane);
    else           wave_gemm32s(ldsW2, ldsS, ldsY, lane);
    __syncthreads();

    unsigned short* dst = out + (((size_t)t * B_ + b) * C_ + c) * HW_;
    for (int j = 0; j < 32; ++j) {         // bias + bn2 + residual + LIF2
      int idx = lane + j * 32;
      int m = idx >> 5, n = idx & 31;
      float bias = (MODE == 0) ? b2[n] : b2[m];
      float y = (ldsY[idx] + bias) * s2 + t2 + ldsH[idx];
      float v = 0.5f * (v2s[idx] + y);
      bool  s = (v >= 1.0f);
      v2s[idx] = s ? 0.0f : v;
      dst[idx] = s ? (unsigned short)0x3F80 : (unsigned short)0;
    }
    __syncthreads();
  }
}

// ---------------------------------------------------------------------------
// Kernel C: out[t,b,o,hw] = sum_c W[o,c] * xcat[c,hw] + bias[o]
// Per (t,b): GEMM M=256, N=1024, K=768 (virtual concat of xh|xw|xid).
// Workgroup tile 64(M) x 128(N); 8 waves, 4 accumulator tiles each;
// 24 K-steps of 32 via v_wmma_f32_16x16x32_bf16, operands swizzled in LDS.
// ---------------------------------------------------------------------------
__global__ __launch_bounds__(256) void proj_kernel(
    const unsigned short* __restrict__ xh,
    const unsigned short* __restrict__ xw,
    const unsigned short* __restrict__ xid,
    const float* __restrict__ Wp, const float* __restrict__ bp,
    float* __restrict__ out)
{
  __shared__ unsigned short Abuf[64 * 32];   // 4 swizzled 16-row tiles
  __shared__ unsigned short Bbuf[32 * 128];  // 8 swizzled 16-col tiles

  const int tb   = blockIdx.y;
  const int tile = blockIdx.x;
  const int mo   = (tile >> 3) * 64;
  const int n0   = (tile & 7) * 128;
  const int tid  = threadIdx.x;
  const int lane = tid & 31;
  const int wid  = tid >> 5;
  const int wm   = wid >> 1, wn = wid & 1;
  const int half = lane >> 4, mn = lane & 15;
  const size_t planeBase = (size_t)tb * C_ * HW_;

  v8f z = {0.f, 0.f, 0.f, 0.f, 0.f, 0.f, 0.f, 0.f};
  v8f acc[4] = {z, z, z, z};

  for (int kk = 0; kk < 24; ++kk) {
    const int k0 = kk * 32;
    if (kk + 1 < 24)   // pull next weight tile toward the caches
      __builtin_prefetch(&Wp[(size_t)mo * 768 + k0 + 32], 0, 0);

    // A: 64x32 of out_w (f32 -> bf16), float4 loads, swizzled scatter
#pragma unroll
    for (int r = 0; r < 2; ++r) {
      int e4 = tid + r * 256;                // 512 float4 groups
      int o = e4 >> 3, ckg = (e4 & 7) * 4;
      f4 v = *(const f4*)&Wp[(size_t)(mo + o) * 768 + k0 + ckg];
#pragma unroll
      for (int e = 0; e < 4; ++e)
        Abuf[swzA(o, ckg + e)] = f2bf(v[e]);
    }
    // B: 32x128 of the spike concat, ushort8 loads, swizzled scatter
#pragma unroll
    for (int r = 0; r < 2; ++r) {
      int e8 = tid + r * 256;                // 512 ushort8 groups
      int ck = e8 >> 4, n8 = (e8 & 15) * 8;
      int gc = k0 + ck;
      const unsigned short* src = (gc < 256) ? xh : (gc < 512) ? xw : xid;
      us8 v = *(const us8*)&src[planeBase + (size_t)(gc & 255) * HW_ + n0 + n8];
#pragma unroll
      for (int e = 0; e < 8; ++e)
        Bbuf[swzB(ck, n8 + e)] = v[e];
    }
    __syncthreads();

    v16bf a = *(const v16bf*)&Abuf[wm * 512 + lane * 16];
#pragma unroll
    for (int tj = 0; tj < 4; ++tj) {
      v16bf bb = *(const v16bf*)&Bbuf[(wn * 4 + tj) * 512 + lane * 16];
      acc[tj] = __builtin_amdgcn_wmma_f32_16x16x32_bf16(false, a, false, bb,
                                                        (short)0, acc[tj],
                                                        false, false);
    }
    __syncthreads();
  }

#pragma unroll
  for (int tj = 0; tj < 4; ++tj)
#pragma unroll
    for (int r = 0; r < 8; ++r) {
      int m = mo + wm * 16 + half * 8 + r;
      int n = n0 + wn * 64 + tj * 16 + mn;
      out[planeBase + (size_t)m * HW_ + n] = acc[tj][r] + bp[m];
    }
}

// ---------------------------------------------------------------------------
// Launch. Input order = setup_inputs() dict insertion order, flattened:
//  0 feature_map
//  1..4  pre_bn (g,b,m,v)
//  5 dw_w   6 dw_b   7..10 dw_bn (g,b,m,v)
//  11 mlp_h.w1 12 b1 13..16 bn1 17 w2 18 b2 19..22 bn2
//  23 mlp_w.w1 24 b1 25..28 bn1 29 w2 30 b2 31..34 bn2
//  35 out_w 36 out_b
// ---------------------------------------------------------------------------
extern "C" void kernel_launch(void* const* d_in, const int* in_sizes, int n_in,
                              void* d_out, int out_size, void* d_ws, size_t ws_size,
                              hipStream_t stream) {
  const float* fm = (const float*)d_in[0];
  const float* pre_g = (const float*)d_in[1];
  const float* pre_b = (const float*)d_in[2];
  const float* pre_m = (const float*)d_in[3];
  const float* pre_v = (const float*)d_in[4];
  const float* dw_w  = (const float*)d_in[5];
  const float* dw_b  = (const float*)d_in[6];
  const float* dbn_g = (const float*)d_in[7];
  const float* dbn_b = (const float*)d_in[8];
  const float* dbn_m = (const float*)d_in[9];
  const float* dbn_v = (const float*)d_in[10];
  const float* out_w = (const float*)d_in[35];
  const float* out_b = (const float*)d_in[36];

  const size_t spkN = (size_t)T_ * B_ * C_ * HW_;   // elements
  unsigned short* xspk = (unsigned short*)d_ws;     // x_id spikes
  unsigned short* xh   = xspk + spkN;               // mlp_h spikes
  unsigned short* xw   = xh + spkN;                 // mlp_w spikes

  lif_front_kernel<<<B_ * C_, 256, 0, stream>>>(
      fm, pre_g, pre_b, pre_m, pre_v, dw_w, dw_b,
      dbn_g, dbn_b, dbn_m, dbn_v, xspk);

  // mlp_h (MODE 1: Y = W @ S, mixing along H)
  mlp_kernel<1><<<B_ * C_, 32, 0, stream>>>(
      xspk,
      (const float*)d_in[11], (const float*)d_in[12],
      (const float*)d_in[13], (const float*)d_in[14],
      (const float*)d_in[15], (const float*)d_in[16],
      (const float*)d_in[17], (const float*)d_in[18],
      (const float*)d_in[19], (const float*)d_in[20],
      (const float*)d_in[21], (const float*)d_in[22],
      xh);

  // mlp_w (MODE 0: Y = S @ W^T, mixing along W)
  mlp_kernel<0><<<B_ * C_, 32, 0, stream>>>(
      xspk,
      (const float*)d_in[23], (const float*)d_in[24],
      (const float*)d_in[25], (const float*)d_in[26],
      (const float*)d_in[27], (const float*)d_in[28],
      (const float*)d_in[29], (const float*)d_in[30],
      (const float*)d_in[31], (const float*)d_in[32],
      (const float*)d_in[33], (const float*)d_in[34],
      xw);

  proj_kernel<<<dim3(32, T_ * B_), 256, 0, stream>>>(
      xh, xw, xspk, out_w, out_b, (float*)d_out);
}
